// GlobalAdaptKernelModule_19542101196950
// MI455X (gfx1250) — matrix operand order repaired and checked by reading, not compiled, not run on hardware
//
#include <hip/hip_runtime.h>

// ---------------------------------------------------------------------------
// CDNA5 (gfx1250): bf16 WMMA GEMMs + async global->LDS staging
// ---------------------------------------------------------------------------

typedef __attribute__((ext_vector_type(16))) __bf16        bf16x16;
typedef __attribute__((ext_vector_type(8)))  float         f32x8;
typedef __attribute__((ext_vector_type(4)))  unsigned int  u32x4;

union FragU { bf16x16 v; u32x4 q[2]; };

#define NB      32
#define CIN     512
#define CM      128
#define POUT    512
#define HDIM    56
#define HW      3136          // 56*56
#define PADW    58
#define PADHW   3364          // 58*58
#define PTILE   112           // 2 rows of 56; 7 x 16-col WMMA subtiles
#define NTILES  28            // 3136 / 112
#define RS      40            // stage-B LDS row stride (ushorts) = 80B, 16B aligned
#define RSC     136           // stage-C/D LDS row stride (ushorts) = 272B, 16B aligned

// round-to-nearest-even fp32 -> bf16 (raw ushort)
__device__ __forceinline__ unsigned int f2bf_u32(float f) {
  unsigned int u = __float_as_uint(f);
  u += 0x7fffu + ((u >> 16) & 1u);
  return u >> 16;
}
__device__ __forceinline__ unsigned short f2bf(float f) {
  return (unsigned short)f2bf_u32(f);
}
// pack two floats -> two bf16 in one dword (lo = first elem)
__device__ __forceinline__ unsigned int f2bf_pk(float lo, float hi) {
  return f2bf_u32(lo) | (f2bf_u32(hi) << 16);
}

__device__ __forceinline__ f32x8 wmma_bf16(const FragU& a, const FragU& b, f32x8 c) {
  return __builtin_amdgcn_wmma_f32_16x16x32_bf16(false, a.v, false, b.v,
                                                 (short)0, c, false, false);
}

// A fragment: row-major bf16, lane row = (lane&15), K segs at +0 / +16 elems.
__device__ __forceinline__ void loadA(FragU& f, const unsigned short* p) {
  f.q[0] = *(const u32x4*)(p);
  f.q[1] = *(const u32x4*)(p + 16);
}
// B fragment: lane column, K segs at +0 / +16 elems.
__device__ __forceinline__ void loadB(FragU& f, const unsigned short* p) {
  f.q[0] = *(const u32x4*)(p);
  f.q[1] = *(const u32x4*)(p + 16);
}

// async 16B global->LDS copy (per-lane), tracked by ASYNCcnt
__device__ __forceinline__ void async_cp16(unsigned lds_off, const void* gaddr) {
  asm volatile("global_load_async_to_lds_b128 %0, %1, off"
               :: "v"(lds_off), "v"(gaddr) : "memory");
}
__device__ __forceinline__ void wait_async0() {
  asm volatile("s_wait_asynccnt 0x0" ::: "memory");
}

// ---------------------------------------------------------------------------
// small prep kernels
// ---------------------------------------------------------------------------

__global__ void zero_u128(u32x4* p, long long n) {
  u32x4 z = {0u, 0u, 0u, 0u};
  for (long long i = blockIdx.x * (long long)blockDim.x + threadIdx.x; i < n;
       i += (long long)gridDim.x * blockDim.x)
    p[i] = z;
}

__global__ void prep_weights(const float* conv_w, const float* fuse_w,
                             unsigned int* convw_bf, unsigned int* fusew_bf) {
  int i = blockIdx.x * blockDim.x + threadIdx.x;              // dword pairs
  for (; i < CM * CIN / 2; i += gridDim.x * blockDim.x) {
    convw_bf[i] = f2bf_pk(conv_w[2 * i], conv_w[2 * i + 1]);
    fusew_bf[i] = f2bf_pk(fuse_w[2 * i], fuse_w[2 * i + 1]);
  }
}

__global__ __launch_bounds__(128) void pool_kernel(const float* x, float* pooled) {
  __shared__ float red[128];
  const float* base = x + (size_t)blockIdx.x * HW;
  float s = 0.f;
  for (int i = threadIdx.x; i < HW; i += 128) s += base[i];
  red[threadIdx.x] = s;
  __syncthreads();
  for (int off = 64; off > 0; off >>= 1) {
    if (threadIdx.x < off) red[threadIdx.x] += red[threadIdx.x + off];
    __syncthreads();
  }
  if (threadIdx.x == 0) pooled[blockIdx.x] = red[0] * (1.0f / (float)HW);
}

__global__ __launch_bounds__(128) void g_kernel(const float* pooled, const float* conv_w,
                                                const float* conv_b, float* g) {
  __shared__ float sp[CIN];
  int n = blockIdx.x, m = threadIdx.x;
  for (int c = threadIdx.x; c < CIN; c += 128) sp[c] = pooled[n * CIN + c];
  __syncthreads();
  float s = conv_b[m];
  const float* wr = conv_w + (size_t)m * CIN;
  for (int c = 0; c < CIN; ++c) s += sp[c] * wr[c];
  g[n * CM + m] = s > 0.f ? s : 0.f;
}

// Wk_bf16[n][kk][oc][ic]: one dword (2 consecutive ic) per thread iteration
__global__ void wk_kernel(const float* g, const float* ck_w, const float* ck_b,
                          const float* ckk_w, const float* ckk_b, unsigned int* wk) {
  const int total = NB * 9 * CM * CM / 2;
  for (int idx = blockIdx.x * blockDim.x + threadIdx.x; idx < total;
       idx += gridDim.x * blockDim.x) {
    int ic2 = (idx & 63) * 2;
    int r = idx >> 6;
    int oc = r & 127;
    r >>= 7;
    int kk = r % 9;
    int n = r / 9;
    float cw = ck_w[kk], cb = ck_b[kk];
    float kw = ckk_w[oc], kb = ckk_b[oc];
    float v0 = kw * (g[n * CM + ic2] * cw + cb) + kb;
    float v1 = kw * (g[n * CM + ic2 + 1] * cw + cb) + kb;
    wk[idx] = f2bf_pk(v0 > 0.f ? v0 : 0.f, v1 > 0.f ? v1 : 0.f);
  }
}

// ---------------------------------------------------------------------------
// Stage B: f = relu(conv_w @ x[n]) -> padded transposed bf16 f_pad_t[n][ppad][ic]
// software-pipelined: next chunk's float4 global loads overlap current WMMAs
// ---------------------------------------------------------------------------
__global__ __launch_bounds__(256) void front_gemm(const float* __restrict__ x,
                                                  const unsigned short* __restrict__ wbf,
                                                  const float* __restrict__ conv_b,
                                                  unsigned short* __restrict__ fpad) {
  __shared__ __align__(16) unsigned short Bt[PTILE * RS];
  const int tile = blockIdx.x, n = blockIdx.y;
  const int p0 = tile * PTILE;
  const int tid = threadIdx.x, lane = tid & 31, wid = tid >> 5;
  const int lrow = lane & 15, lhi = lane >> 4;

  f32x8 zero = {0.f, 0.f, 0.f, 0.f, 0.f, 0.f, 0.f, 0.f};
  f32x8 acc[7];
#pragma unroll
  for (int t = 0; t < 7; ++t) acc[t] = zero;

  const float* xn = x + (size_t)n * CIN * HW + p0;

  // staged chunk: 32 c x 112 pos = 896 float4 segments; clamp avoids exec-mask
  // divergence (segment 895 is duplicated with identical data - benign)
  int segi[4];
#pragma unroll
  for (int i = 0; i < 4; ++i) {
    int s = tid + i * 256;
    segi[i] = s < 896 ? s : 895;
  }

  float4 xv[4];
#pragma unroll
  for (int i = 0; i < 4; ++i) {
    int c = segi[i] / 28, p4 = segi[i] % 28;
    xv[i] = *(const float4*)(xn + (size_t)c * HW + p4 * 4);
  }

  for (int kc = 0; kc < CIN; kc += 32) {
    __syncthreads();                       // prior chunk's LDS reads complete
#pragma unroll
    for (int i = 0; i < 4; ++i) {
      int c = segi[i] / 28, p = (segi[i] % 28) * 4;
      Bt[(p + 0) * RS + c] = f2bf(xv[i].x);
      Bt[(p + 1) * RS + c] = f2bf(xv[i].y);
      Bt[(p + 2) * RS + c] = f2bf(xv[i].z);
      Bt[(p + 3) * RS + c] = f2bf(xv[i].w);
    }
    __syncthreads();                       // Bt ready
    if (kc + 32 < CIN) {                   // prefetch next chunk (overlaps WMMAs)
#pragma unroll
      for (int i = 0; i < 4; ++i) {
        int c = segi[i] / 28, p4 = segi[i] % 28;
        xv[i] = *(const float4*)(xn + (size_t)(kc + 32 + c) * HW + p4 * 4);
      }
    }

    FragU fa;
    loadA(fa, wbf + (size_t)(16 * wid + lrow) * CIN + kc + lhi * 8);
    FragU fb[7];
#pragma unroll
    for (int t = 0; t < 7; ++t)
      loadB(fb[t], &Bt[(t * 16 + lrow) * RS + lhi * 8]);
#pragma unroll
    for (int t = 0; t < 7; ++t)
      acc[t] = wmma_bf16(fa, fb[t], acc[t]);
  }

  // epilogue: relu(acc + conv_b), packed pair stores (m, m+1 contiguous, even)
#pragma unroll
  for (int t = 0; t < 7; ++t) {
    int pos = p0 + t * 16 + lrow;
    int h = pos / HDIM, w = pos % HDIM;
    size_t prow = ((size_t)n * PADHW + (size_t)(h + 1) * PADW + (w + 1)) * CM;
#pragma unroll
    for (int r = 0; r < 8; r += 2) {
      int m = 16 * wid + r + 8 * lhi;
      float v0 = acc[t][r] + conv_b[m];
      float v1 = acc[t][r + 1] + conv_b[m + 1];
      v0 = v0 > 0.f ? v0 : 0.f;
      v1 = v1 > 0.f ? v1 : 0.f;
      *(unsigned int*)(fpad + prow + m) = f2bf_pk(v0, v1);
    }
  }
}

// ---------------------------------------------------------------------------
// Stage C: y[n] = dynconv3x3(f[n], Wk[n]) + adap_b -> y_t[n][pos][oc] bf16
// All 9 taps for a 2-row tile touch exactly padded rows 2T..2T+3 x 58 cols:
// one contiguous 58KB chunk -> staged ONCE per block via async global->LDS.
// ---------------------------------------------------------------------------
__global__ __launch_bounds__(256) void dyn_conv(const unsigned short* __restrict__ fpad,
                                                const unsigned short* __restrict__ wk,
                                                const float* __restrict__ adap_b,
                                                unsigned short* __restrict__ yt) {
  __shared__ __align__(16) unsigned short Lt[4 * PADW * RSC];   // 232 rows * 136
  const int tile = blockIdx.x, n = blockIdx.y;
  const int p0 = tile * PTILE;
  const int tid = threadIdx.x, lane = tid & 31, wid = tid >> 5;
  const int lrow = lane & 15, lhi = lane >> 4;

  // async-stage 232 padded positions x 128 ic (contiguous in global)
  {
    const unsigned short* gbase =
        fpad + ((size_t)n * PADHW + (size_t)(2 * tile) * PADW) * CM;
    unsigned lbase = (unsigned)(size_t)(void*)Lt;
    for (int s = tid; s < 4 * PADW * 16; s += 256) {      // 3712 x 16B segs
      int row = s >> 4, seg = s & 15;
      async_cp16(lbase + (unsigned)(row * RSC + seg * 8) * 2, gbase + s * 8);
    }
  }

  f32x8 zero = {0.f, 0.f, 0.f, 0.f, 0.f, 0.f, 0.f, 0.f};
  f32x8 acc[7];
#pragma unroll
  for (int t = 0; t < 7; ++t) acc[t] = zero;

  // per-lane padded row index within the 4x58 LDS tile (before tap shift)
  int prow[7];
#pragma unroll
  for (int t = 0; t < 7; ++t) {
    int pl = t * 16 + lrow;                 // 0..111
    int hi = (pl >= HDIM) ? 1 : 0;
    int w = pl - HDIM * hi;
    prow[t] = (1 + hi) * PADW + (w + 1);
  }

  const unsigned short* wkn = wk + (size_t)n * 9 * CM * CM;

  wait_async0();
  __syncthreads();

  for (int kk = 0; kk < 9; ++kk) {
    const int dh = kk / 3 - 1, dw = kk % 3 - 1;
    const int shift = dh * PADW + dw;
    const unsigned short* wrow =
        wkn + (size_t)kk * CM * CM + (size_t)(16 * wid + lrow) * CM + lhi * 8;
#pragma unroll
    for (int icc = 0; icc < 4; ++icc) {
      FragU fa;
      loadA(fa, wrow + icc * 32);
      FragU fb[7];
#pragma unroll
      for (int t = 0; t < 7; ++t)
        loadB(fb[t], &Lt[(prow[t] + shift) * RSC + icc * 32 + lhi * 8]);
#pragma unroll
      for (int t = 0; t < 7; ++t)
        acc[t] = wmma_bf16(fa, fb[t], acc[t]);
    }
  }

  // epilogue: y = acc + adap_b (no relu), packed pair stores (oc even-aligned)
#pragma unroll
  for (int t = 0; t < 7; ++t) {
    int pos = p0 + t * 16 + lrow;
    size_t prow_o = ((size_t)n * HW + pos) * CM;
#pragma unroll
    for (int r = 0; r < 8; r += 2) {
      int oc = 16 * wid + r + 8 * lhi;
      *(unsigned int*)(yt + prow_o + oc) =
          f2bf_pk(acc[t][r] + adap_b[oc], acc[t][r + 1] + adap_b[oc + 1]);
    }
  }
}

// ---------------------------------------------------------------------------
// Stage D: out[n] = fuse_w @ y[n] + fuse_b ; y tile async-staged once (28KB)
// ---------------------------------------------------------------------------
__global__ __launch_bounds__(256) void fuse_gemm(const unsigned short* __restrict__ yt,
                                                 const unsigned short* __restrict__ fw,
                                                 const float* __restrict__ fuse_b,
                                                 float* __restrict__ out) {
  __shared__ __align__(16) unsigned short Yt[PTILE * RSC];      // 112 * 136
  const int tile = blockIdx.x, n = blockIdx.y, mb = blockIdx.z;
  const int p0 = tile * PTILE;
  const int tid = threadIdx.x, lane = tid & 31, wid = tid >> 5;
  const int lrow = lane & 15, lhi = lane >> 4;

  {
    const unsigned short* gbase = yt + ((size_t)n * HW + p0) * CM;
    unsigned lbase = (unsigned)(size_t)(void*)Yt;
    for (int s = tid; s < PTILE * 16; s += 256) {         // 1792 x 16B segs
      int row = s >> 4, seg = s & 15;
      async_cp16(lbase + (unsigned)(row * RSC + seg * 8) * 2, gbase + s * 8);
    }
  }

  f32x8 zero = {0.f, 0.f, 0.f, 0.f, 0.f, 0.f, 0.f, 0.f};
  f32x8 acc[7];
#pragma unroll
  for (int t = 0; t < 7; ++t) acc[t] = zero;

  const unsigned short* arow =
      fw + (size_t)(mb * 128 + 16 * wid + lrow) * CM + lhi * 8;

  wait_async0();
  __syncthreads();

#pragma unroll
  for (int kc = 0; kc < CM; kc += 32) {
    FragU fa;
    loadA(fa, arow + kc);
    FragU fb[7];
#pragma unroll
    for (int t = 0; t < 7; ++t)
      loadB(fb[t], &Yt[(t * 16 + lrow) * RSC + kc + lhi * 8]);
#pragma unroll
    for (int t = 0; t < 7; ++t)
      acc[t] = wmma_bf16(fa, fb[t], acc[t]);
  }

#pragma unroll
  for (int t = 0; t < 7; ++t) {
    int pos = p0 + t * 16 + lrow;
#pragma unroll
    for (int r = 0; r < 8; ++r) {
      int p = mb * 128 + 16 * wid + r + 8 * lhi;
      out[((size_t)n * POUT + p) * HW + pos] = acc[t][r] + fuse_b[p];
    }
  }
}

// ---------------------------------------------------------------------------
// launch
// ---------------------------------------------------------------------------
extern "C" void kernel_launch(void* const* d_in, const int* in_sizes, int n_in,
                              void* d_out, int out_size, void* d_ws, size_t ws_size,
                              hipStream_t stream) {
  (void)in_sizes; (void)n_in; (void)out_size; (void)ws_size;
  const float* x      = (const float*)d_in[0];
  const float* conv_w = (const float*)d_in[1];
  const float* conv_b = (const float*)d_in[2];
  const float* ck_w   = (const float*)d_in[3];
  const float* ck_b   = (const float*)d_in[4];
  const float* ckk_w  = (const float*)d_in[5];
  const float* ckk_b  = (const float*)d_in[6];
  const float* adap_b = (const float*)d_in[7];
  const float* fuse_w = (const float*)d_in[8];
  const float* fuse_b = (const float*)d_in[9];
  float* out = (float*)d_out;

  char* ws = (char*)d_ws;
  const size_t OFF_CONVW = 0;
  const size_t OFF_FUSEW = OFF_CONVW + (size_t)CM * CIN * 2;
  const size_t OFF_POOL  = OFF_FUSEW + (size_t)POUT * CM * 2;
  const size_t OFF_G     = OFF_POOL + (size_t)NB * CIN * 4;
  const size_t OFF_WK    = OFF_G + (size_t)NB * CM * 4;
  const size_t OFF_FPAD  = OFF_WK + (size_t)NB * 9 * CM * CM * 2;
  const size_t OFF_YT    = OFF_FPAD + (size_t)NB * PADHW * CM * 2;

  unsigned short* convw_bf = (unsigned short*)(ws + OFF_CONVW);
  unsigned short* fusew_bf = (unsigned short*)(ws + OFF_FUSEW);
  float*          pooled   = (float*)(ws + OFF_POOL);
  float*          g        = (float*)(ws + OFF_G);
  unsigned short* wkbf     = (unsigned short*)(ws + OFF_WK);
  unsigned short* fpad     = (unsigned short*)(ws + OFF_FPAD);
  unsigned short* ytb      = (unsigned short*)(ws + OFF_YT);

  const long long fpad_v4 = (long long)NB * PADHW * CM * 2 / 16;
  zero_u128<<<2048, 256, 0, stream>>>((u32x4*)fpad, fpad_v4);

  prep_weights<<<128, 256, 0, stream>>>(conv_w, fuse_w,
                                        (unsigned int*)convw_bf,
                                        (unsigned int*)fusew_bf);
  pool_kernel<<<NB * CIN, 128, 0, stream>>>(x, pooled);
  g_kernel<<<NB, 128, 0, stream>>>(pooled, conv_w, conv_b, g);
  wk_kernel<<<2304, 256, 0, stream>>>(g, ck_w, ck_b, ckk_w, ckk_b,
                                      (unsigned int*)wkbf);

  front_gemm<<<dim3(NTILES, NB), 256, 0, stream>>>(x, convw_bf, conv_b, fpad);
  dyn_conv<<<dim3(NTILES, NB), 256, 0, stream>>>(fpad, wkbf, adap_b, ytb);
  fuse_gemm<<<dim3(NTILES, NB, 4), 256, 0, stream>>>(ytb, fusew_bf, fuse_b, out);
}
